// LinearAttention_45200235823734
// MI455X (gfx1250) — matrix operand order repaired and checked by reading, not compiled, not run on hardware
//
#include <hip/hip_runtime.h>
#include <hip/hip_bf16.h>
#include <math.h>

// ---------------- problem constants ----------------
#define DIM        2048
#define N_HEADS    16
#define N_KV_HEADS 8
#define HEAD_DIM   128
#define N_REP      2
#define BATCH      2
#define SEQ        2048
#define CHUNK      128
#define ROWS       (BATCH*SEQ)  // 4096

// ---------------- CDNA5 async-to-LDS plumbing ----------------
#ifndef __has_builtin
#define __has_builtin(x) 0
#endif
#if __has_builtin(__builtin_amdgcn_global_load_async_to_lds_b128) && \
    __has_builtin(__builtin_amdgcn_s_wait_asynccnt)
#define HAVE_ASYNC_LDS 1
#else
#define HAVE_ASYNC_LDS 0
#endif

typedef __attribute__((ext_vector_type(4))) int i32x4;

// copy 16 bytes global -> LDS (async on gfx1250 when available)
__device__ __forceinline__ void copy16_to_lds(const _Float16* g, _Float16* l) {
#if HAVE_ASYNC_LDS
  __builtin_amdgcn_global_load_async_to_lds_b128(
      (__attribute__((address_space(1))) i32x4*)g,
      (__attribute__((address_space(3))) i32x4*)l, 0, 0);
#else
  *(uint4*)l = *(const uint4*)g;
#endif
}
__device__ __forceinline__ void async_wait() {
#if HAVE_ASYNC_LDS
  __builtin_amdgcn_s_wait_asynccnt(0);
#endif
}

// ---------------- WMMA plumbing (CDNA5, wave32) ----------------
typedef __attribute__((ext_vector_type(16))) _Float16 v16h;
typedef __attribute__((ext_vector_type(8)))  float    v8f;

union F16Frag { v16h v; _Float16 e[16]; };

__device__ __forceinline__ v8f wmma_f16(v16h a, v16h b, v8f c) {
  return __builtin_amdgcn_wmma_f32_16x16x32_f16(
      false, a, false, b, (short)0, c, false, false);
}

// A-matrix fragment (16x32 f16), element s holds k = 16*(lane/16) + s. Element (m,k) = base[m*sM + k*sK].
__device__ __forceinline__ v16h load_fragA(const _Float16* base, int sM, int sK, int lane) {
  F16Frag f;
  int m  = lane & 15;
  int hi = lane >> 4;
#pragma unroll
  for (int j = 0; j < 8; ++j) {
    int k = hi * 16 + 2 * j;
    f.e[2 * j]     = base[m * sM + k * sK];
    f.e[2 * j + 1] = base[m * sM + (k + 1) * sK];
  }
  return f.v;
}

// B-matrix fragment (32x16 f16). Element (k,n) = base[n*sN + k*sK].
__device__ __forceinline__ v16h load_fragB(const _Float16* base, int sN, int sK, int lane) {
  F16Frag f;
  int n  = lane & 15;
  int hi = lane >> 4;
#pragma unroll
  for (int j = 0; j < 8; ++j) {
    int k = hi * 16 + 2 * j;
    f.e[2 * j]     = base[n * sN + k * sK];
    f.e[2 * j + 1] = base[n * sN + (k + 1) * sK];
  }
  return f.v;
}

// Same as load_fragB but source is f32 LDS (converted on the fly).
__device__ __forceinline__ v16h load_fragB_f32(const float* base, int sN, int sK, int lane) {
  F16Frag f;
  int n  = lane & 15;
  int hi = lane >> 4;
#pragma unroll
  for (int j = 0; j < 8; ++j) {
    int k = hi * 16 + 2 * j;
    f.e[2 * j]     = (_Float16)base[n * sN + k * sK];
    f.e[2 * j + 1] = (_Float16)base[n * sN + (k + 1) * sK];
  }
  return f.v;
}

// ---------------- f32 -> f16 convert (one bandwidth pass) -----------------
__global__ void cvt_f32_f16(const float* __restrict__ src, _Float16* __restrict__ dst,
                            long n) {
  long i = ((long)blockIdx.x * blockDim.x + threadIdx.x) * 4;
  if (i >= n) return;
  float4 f = *(const float4*)(src + i);
  union { _Float16 h[4]; uint2 u; } p;
  p.h[0] = (_Float16)f.x; p.h[1] = (_Float16)f.y;
  p.h[2] = (_Float16)f.z; p.h[3] = (_Float16)f.w;
  *(uint2*)(dst + i) = p.u;
}

// ---------------- GEMM: C[M,N] = A[M,K] * W[N,K]^T  (f16 in, f16 WMMA) ----
// 64x64 block tile, 128 threads = 4 waves (2x2), K-tile 64, async LDS staging.
#define TK2 64
__global__ void gemm_h(const _Float16* __restrict__ A,
                       const _Float16* __restrict__ W,
                       void* __restrict__ Cout,
                       int M, int N, int K, int out_half) {
  __shared__ _Float16 As[64][TK2];
  __shared__ _Float16 Bs[64][TK2];
  const int tid  = threadIdx.x;
  const int lane = tid & 31;
  const int wave = tid >> 5;
  const int wr   = wave >> 1;
  const int wc   = wave & 1;
  const int bm   = blockIdx.y * 64;
  const int bn   = blockIdx.x * 64;

  v8f acc[2][2];
#pragma unroll
  for (int i = 0; i < 2; ++i)
#pragma unroll
    for (int j = 0; j < 2; ++j) acc[i][j] = (v8f){};

  for (int k0 = 0; k0 < K; k0 += TK2) {
    // stage 64x64 f16 tiles: 512 16-byte segments each, 4 per thread
#pragma unroll
    for (int i = 0; i < 4; ++i) {
      int idx = tid + 128 * i;          // 0..511
      int r   = idx >> 3;               // row 0..63
      int c8  = (idx & 7) * 8;          // col segment
      copy16_to_lds(&A[(size_t)(bm + r) * K + k0 + c8], &As[r][c8]);
      copy16_to_lds(&W[(size_t)(bn + r) * K + k0 + c8], &Bs[r][c8]);
    }
    async_wait();
    __syncthreads();
#pragma unroll
    for (int kk = 0; kk < 2; ++kk) {
      int ko = kk * 32;
#pragma unroll
      for (int i = 0; i < 2; ++i) {
        v16h a = load_fragA(&As[wr * 32 + i * 16][ko], TK2, 1, lane);
#pragma unroll
        for (int j = 0; j < 2; ++j) {
          v16h b = load_fragB(&Bs[wc * 32 + j * 16][ko], TK2, 1, lane);
          acc[i][j] = wmma_f16(a, b, acc[i][j]);
        }
      }
    }
    __syncthreads();
  }

  const int n0 = lane & 15;
  const int hi = lane >> 4;
  if (out_half) {
    _Float16* C = (_Float16*)Cout;
#pragma unroll
    for (int i = 0; i < 2; ++i)
#pragma unroll
      for (int j = 0; j < 2; ++j)
#pragma unroll
        for (int r = 0; r < 8; ++r) {
          int m = bm + wr * 32 + i * 16 + r + 8 * hi;
          int n = bn + wc * 32 + j * 16 + n0;
          C[(size_t)m * N + n] = (_Float16)acc[i][j][r];
        }
  } else {
    float* C = (float*)Cout;
#pragma unroll
    for (int i = 0; i < 2; ++i)
#pragma unroll
      for (int j = 0; j < 2; ++j)
#pragma unroll
        for (int r = 0; r < 8; ++r) {
          int m = bm + wr * 32 + i * 16 + r + 8 * hi;
          int n = bn + wc * 32 + j * 16 + n0;
          C[(size_t)m * N + n] = acc[i][j][r];
        }
  }
}

// ---------------- rotary + row softmax for Q (one wave per row, f16 buf) --
__global__ void rope_softmax_q(_Float16* __restrict__ q, const float* __restrict__ freqs) {
  int wid  = blockIdx.x * (blockDim.x >> 5) + (threadIdx.x >> 5);
  int lane = threadIdx.x & 31;
  int h  = wid & (N_HEADS - 1);
  int bs = wid >> 4;
  int s  = bs & (SEQ - 1);
  _Float16* row = q + (size_t)bs * DIM + h * HEAD_DIM;

  float vals[4];
#pragma unroll
  for (int pp = 0; pp < 2; ++pp) {
    int p = lane * 2 + pp;
    float c  = freqs[s * 128 + 2 * p];
    float sn = freqs[s * 128 + 2 * p + 1];
    float x0 = (float)row[2 * p], x1 = (float)row[2 * p + 1];
    vals[2 * pp]     = x0 * c - x1 * sn;
    vals[2 * pp + 1] = x0 * sn + x1 * c;
  }
  const float scale = 0.0883883476483184f;  // 1/sqrt(128)
  float m = -1e30f;
#pragma unroll
  for (int i = 0; i < 4; ++i) { vals[i] *= scale; m = fmaxf(m, vals[i]); }
#pragma unroll
  for (int off = 16; off >= 1; off >>= 1) m = fmaxf(m, __shfl_xor(m, off, 32));
  float sum = 0.f;
#pragma unroll
  for (int i = 0; i < 4; ++i) { vals[i] = __expf(vals[i] - m); sum += vals[i]; }
#pragma unroll
  for (int off = 16; off >= 1; off >>= 1) sum += __shfl_xor(sum, off, 32);
  float inv = 1.f / sum;
#pragma unroll
  for (int pp = 0; pp < 2; ++pp) {
    int p = lane * 2 + pp;
    row[2 * p]     = (_Float16)(vals[2 * pp] * inv);
    row[2 * p + 1] = (_Float16)(vals[2 * pp + 1] * inv);
  }
}

// ---------------- rotary only for K (f16 buf) -----------------------------
__global__ void rope_k(_Float16* __restrict__ k, const float* __restrict__ freqs) {
  int wid  = blockIdx.x * (blockDim.x >> 5) + (threadIdx.x >> 5);
  int lane = threadIdx.x & 31;
  int h  = wid & (N_KV_HEADS - 1);
  int bs = wid >> 3;
  int s  = bs & (SEQ - 1);
  _Float16* row = k + (size_t)bs * (N_KV_HEADS * HEAD_DIM) + h * HEAD_DIM;
#pragma unroll
  for (int pp = 0; pp < 2; ++pp) {
    int p = lane * 2 + pp;
    float c  = freqs[s * 128 + 2 * p];
    float sn = freqs[s * 128 + 2 * p + 1];
    float x0 = (float)row[2 * p], x1 = (float)row[2 * p + 1];
    row[2 * p]     = (_Float16)(x0 * c - x1 * sn);
    row[2 * p + 1] = (_Float16)(x0 * sn + x1 * c);
  }
}

// ---------------- column softmax for K over the sequence axis -------------
__global__ void ksoftmax_cols(_Float16* __restrict__ k) {
  __shared__ float red[256];
  int b = blockIdx.x >> 10;
  int j = blockIdx.x & 1023;
  int tid = threadIdx.x;
  float vals[8];
  float m = -1e30f;
#pragma unroll
  for (int i = 0; i < 8; ++i) {
    int s = tid + 256 * i;
    vals[i] = (float)k[(size_t)(b * SEQ + s) * 1024 + j];
    m = fmaxf(m, vals[i]);
  }
  red[tid] = m; __syncthreads();
  for (int off = 128; off >= 1; off >>= 1) {
    if (tid < off) red[tid] = fmaxf(red[tid], red[tid + off]);
    __syncthreads();
  }
  m = red[0]; __syncthreads();
  float sum = 0.f;
#pragma unroll
  for (int i = 0; i < 8; ++i) { vals[i] = __expf(vals[i] - m); sum += vals[i]; }
  red[tid] = sum; __syncthreads();
  for (int off = 128; off >= 1; off >>= 1) {
    if (tid < off) red[tid] += red[tid + off];
    __syncthreads();
  }
  float inv = 1.f / red[0];
#pragma unroll
  for (int i = 0; i < 8; ++i) {
    int s = tid + 256 * i;
    k[(size_t)(b * SEQ + s) * 1024 + j] = (_Float16)(vals[i] * inv);
  }
}

// ---------------- chunked causal linear attention -------------------------
// One workgroup (8 wave32) per (b,h). KV state (128x128 f32) resident in LDS
// across the 16-chunk scan; q/k/v tiles async-staged as f16; 4 WMMA GEMMs per
// chunk (inter, scores, intra, state-update).
__global__ void attn_chunked(const _Float16* __restrict__ qg,
                             const _Float16* __restrict__ kg,
                             const _Float16* __restrict__ vg,
                             _Float16* __restrict__ outg) {
  extern __shared__ __align__(16) char smem_raw[];
  float*    state = (float*)smem_raw;              // [128][128] f32 (64 KB)
  _Float16* qs = (_Float16*)(state + 128 * 128);   // 32 KB each
  _Float16* ks = qs + 128 * 128;
  _Float16* vs = ks + 128 * 128;
  _Float16* sc = vs + 128 * 128;

  const int bh   = blockIdx.x;
  const int b    = bh / N_HEADS;
  const int h    = bh % N_HEADS;
  const int kvh  = h / N_REP;
  const int tid  = threadIdx.x;
  const int lane = tid & 31;
  const int tr   = tid >> 5;
  const int n0   = lane & 15;
  const int hi   = lane >> 4;

  for (int i = tid; i < 128 * 128; i += 256) state[i] = 0.f;
  __syncthreads();

  for (int nck = 0; nck < SEQ / CHUNK; ++nck) {
    const int s0 = nck * CHUNK;
    // async-stage q/k/v chunk tiles (2048 16B segments per tile, 8/thread)
#pragma unroll
    for (int i = 0; i < 8; ++i) {
      int idx = tid + 256 * i;          // 0..2047
      int r   = idx >> 4;               // row 0..127
      int c8  = (idx & 15) * 8;         // col segment
      size_t srow = (size_t)(b * SEQ + s0 + r);
      copy16_to_lds(&qg[srow * DIM  + h   * HEAD_DIM + c8], &qs[r * 128 + c8]);
      copy16_to_lds(&kg[srow * 1024 + kvh * HEAD_DIM + c8], &ks[r * 128 + c8]);
      copy16_to_lds(&vg[srow * 1024 + kvh * HEAD_DIM + c8], &vs[r * 128 + c8]);
    }
    async_wait();
    __syncthreads();

    // ---- inter = qc @ state (cumulative, exclusive) ----
    v8f acc[8];
#pragma unroll
    for (int t = 0; t < 8; ++t) acc[t] = (v8f){};
#pragma unroll
    for (int kk = 0; kk < 4; ++kk) {
      int k0 = kk * 32;
      v16h a = load_fragA(qs + tr * 16 * 128 + k0, 128, 1, lane);
#pragma unroll
      for (int t = 0; t < 8; ++t) {
        v16h bfr = load_fragB_f32(state + k0 * 128 + t * 16, 1, 128, lane);
        acc[t] = wmma_f16(a, bfr, acc[t]);
      }
    }

    // ---- scores = qc @ kc^T, causal-masked, stored f16 ----
#pragma unroll
    for (int t = 0; t < 8; ++t) {
      v8f s = (v8f){};
#pragma unroll
      for (int kk = 0; kk < 4; ++kk) {
        int k0 = kk * 32;
        v16h a   = load_fragA(qs + tr * 16 * 128 + k0, 128, 1, lane);
        v16h bfr = load_fragB(ks + t * 16 * 128 + k0, 128, 1, lane);
        s = wmma_f16(a, bfr, s);
      }
#pragma unroll
      for (int r = 0; r < 8; ++r) {
        int row = tr * 16 + r + 8 * hi;
        int col = t * 16 + n0;
        float val = (col <= row) ? s[r] : 0.f;
        sc[row * 128 + col] = (_Float16)val;
      }
    }
    __syncthreads();

    // ---- intra: acc += scores @ vc ----
#pragma unroll
    for (int kk = 0; kk < 4; ++kk) {
      int k0 = kk * 32;
      v16h a = load_fragA(sc + tr * 16 * 128 + k0, 128, 1, lane);
#pragma unroll
      for (int t = 0; t < 8; ++t) {
        v16h bfr = load_fragB(vs + k0 * 128 + t * 16, 1, 128, lane);
        acc[t] = wmma_f16(a, bfr, acc[t]);
      }
    }

    // ---- write out (inter+intra) as f16 ----
#pragma unroll
    for (int t = 0; t < 8; ++t)
#pragma unroll
      for (int r = 0; r < 8; ++r) {
        int row = tr * 16 + r + 8 * hi;
        int col = t * 16 + n0;
        outg[(size_t)(b * SEQ + s0 + row) * DIM + h * HEAD_DIM + col] =
            (_Float16)acc[t][r];
      }

    // ---- state += kc^T @ vc ----
#pragma unroll
    for (int t = 0; t < 8; ++t) {
      v8f kv = (v8f){};
#pragma unroll
      for (int kk = 0; kk < 4; ++kk) {
        int k0 = kk * 32;
        v16h a   = load_fragA(ks + k0 * 128 + tr * 16, 1, 128, lane);
        v16h bfr = load_fragB(vs + k0 * 128 + t * 16, 1, 128, lane);
        kv = wmma_f16(a, bfr, kv);
      }
#pragma unroll
      for (int r = 0; r < 8; ++r) {
        int row = tr * 16 + r + 8 * hi;
        int col = t * 16 + n0;
        state[row * 128 + col] += kv[r];
      }
    }
    __syncthreads();
  }
}

// ---------------- launcher ------------------------------------------------
extern "C" void kernel_launch(void* const* d_in, const int* in_sizes, int n_in,
                              void* d_out, int out_size, void* d_ws, size_t ws_size,
                              hipStream_t stream) {
  (void)in_sizes; (void)n_in; (void)out_size; (void)ws_size;
  const float* x     = (const float*)d_in[0];
  const float* freqs = (const float*)d_in[1];
  const float* wq    = (const float*)d_in[2];
  const float* wk    = (const float*)d_in[3];
  const float* wv    = (const float*)d_in[4];
  const float* wo    = (const float*)d_in[5];
  float* out = (float*)d_out;

  // f16 workspace layout
  _Float16* xh    = (_Float16*)d_ws;                 // 8M
  _Float16* wqh   = xh   + (size_t)ROWS * DIM;       // 4M
  _Float16* wkh   = wqh  + (size_t)DIM * DIM;        // 2M
  _Float16* wvh   = wkh  + (size_t)1024 * DIM;       // 2M
  _Float16* woh   = wvh  + (size_t)1024 * DIM;       // 4M
  _Float16* qh    = woh  + (size_t)DIM * DIM;        // 8M
  _Float16* kh    = qh   + (size_t)ROWS * DIM;       // 4M
  _Float16* vh    = kh   + (size_t)ROWS * 1024;      // 4M
  _Float16* attnh = vh   + (size_t)ROWS * 1024;      // 8M

  // one-pass f32 -> f16 conversion of x and all weights
  cvt_f32_f16<<<(ROWS * (long)DIM) / 1024, 256, 0, stream>>>(x,  xh,  (long)ROWS * DIM);
  cvt_f32_f16<<<((long)DIM * DIM)  / 1024, 256, 0, stream>>>(wq, wqh, (long)DIM * DIM);
  cvt_f32_f16<<<((long)1024 * DIM) / 1024, 256, 0, stream>>>(wk, wkh, (long)1024 * DIM);
  cvt_f32_f16<<<((long)1024 * DIM) / 1024, 256, 0, stream>>>(wv, wvh, (long)1024 * DIM);
  cvt_f32_f16<<<((long)DIM * DIM)  / 1024, 256, 0, stream>>>(wo, woh, (long)DIM * DIM);

  dim3 blk(128);
  // QKV projections (f16 out)
  gemm_h<<<dim3(DIM / 64,  ROWS / 64), blk, 0, stream>>>(xh, wqh, qh, ROWS, DIM,  DIM, 1);
  gemm_h<<<dim3(1024 / 64, ROWS / 64), blk, 0, stream>>>(xh, wkh, kh, ROWS, 1024, DIM, 1);
  gemm_h<<<dim3(1024 / 64, ROWS / 64), blk, 0, stream>>>(xh, wvh, vh, ROWS, 1024, DIM, 1);
  // rotary + softmaxes
  rope_softmax_q<<<(ROWS * N_HEADS) / 4, 128, 0, stream>>>(qh, freqs);
  rope_k        <<<(ROWS * N_KV_HEADS) / 4, 128, 0, stream>>>(kh, freqs);
  ksoftmax_cols <<<BATCH * 1024, 256, 0, stream>>>(kh);
  // chunked causal linear attention (192 KB LDS: f32 state + 4 f16 tiles)
  size_t smem = 128 * 128 * sizeof(float) + 4 * 128 * 128 * sizeof(_Float16);
  attn_chunked<<<BATCH * N_HEADS, 256, smem, stream>>>(qh, kh, vh, attnh);
  // output projection (f32 out to d_out)
  gemm_h<<<dim3(DIM / 64, ROWS / 64), blk, 0, stream>>>(attnh, woh, out, ROWS, DIM, DIM, 0);
}
